// LSTMModel_11879879542910
// MI455X (gfx1250) — matrix operand order repaired and checked by reading, not compiled, not run on hardware
//
#include <hip/hip_runtime.h>

// LSTM (B=4096, T=512, I=2, H=32) + FC(32->16 tanh) + FC(16->1), fully fused.
// One wave32 per 16-row batch tile (256 waves total), state held in registers,
// recurrent GEMM and fc1 done with v_wmma_f32_16x16x32_f16 (f32 accumulate).

#define B_SZ 4096
#define T_SZ 512

typedef _Float16 v16h  __attribute__((ext_vector_type(16)));
typedef float    v8f   __attribute__((ext_vector_type(8)));
typedef _Float16 h2_t  __attribute__((ext_vector_type(2)));

union V16H {
  v16h v;
  h2_t p[8];
};

__device__ __forceinline__ h2_t pk2(float a, float b) {
  // builtin returns __fp16 ext_vector(2); identical layout to _Float16 vec
  return __builtin_bit_cast(h2_t, __builtin_amdgcn_cvt_pkrtz(a, b));
}

__device__ __forceinline__ float fexp2(float x) {
#if __has_builtin(__builtin_amdgcn_exp2f)
  return __builtin_amdgcn_exp2f(x);          // v_exp_f32
#else
  return exp2f(x);
#endif
}
__device__ __forceinline__ float frcp(float x) {
#if __has_builtin(__builtin_amdgcn_rcpf)
  return __builtin_amdgcn_rcpf(x);           // v_rcp_f32
#else
  return 1.0f / x;
#endif
}
__device__ __forceinline__ float sigmoid_fast(float x) {
  return frcp(1.0f + fexp2(-1.4426950408889634f * x));
}
__device__ __forceinline__ float tanh_fast(float x) {
#if __has_builtin(__builtin_amdgcn_tanhf)
  return __builtin_amdgcn_tanhf(x);          // v_tanh_f32 (gfx1250 TRANS op)
#else
  return 2.0f * frcp(1.0f + fexp2(-2.8853900817779268f * x)) - 1.0f;
#endif
}

__global__ __launch_bounds__(32) void lstm_fused_wmma(
    const float* __restrict__ x,      // [B, T, 2]
    const float* __restrict__ W_ih,   // [128, 2]
    const float* __restrict__ W_hh,   // [128, 32]
    const float* __restrict__ b_ih,   // [128]
    const float* __restrict__ b_hh,   // [128]
    const float* __restrict__ W1,     // [16, 32]
    const float* __restrict__ b1,     // [16]
    const float* __restrict__ W2,     // [1, 16]
    const float* __restrict__ b2,     // [1]
    float* __restrict__ out)          // [B, T]
{
  __shared__ float lds_h[16 * 32];    // h   in row-major [m][k], wave-private
  __shared__ float lds_th[16 * 32];   // tanh(h)

  const int lane = threadIdx.x;       // 0..31 (wave32)
  const int n    = lane & 15;         // column index inside tile
  const int hh   = lane >> 4;         // lane-half
  const int b0   = blockIdx.x * 16;   // batch tile origin

  // ---- W_hh as 8 B-matrix tiles (32x16 f16 each), loaded once ----
  // B layout: lane n = col, VGPR v holds K = hh*16 + 2v, 2v+1
  V16H bhh[8];
#pragma unroll
  for (int ct = 0; ct < 8; ++ct) {
#pragma unroll
    for (int v = 0; v < 8; ++v) {
      const int K = hh * 16 + 2 * v;
      const float* s = W_hh + (ct * 16 + n) * 32 + K;  // B[k][n] = W_hh[16ct+n][k]
      bhh[ct].p[v] = pk2(s[0], s[1]);
    }
  }
  // ---- W1 as one B tile (32x16 f16): B[k][d] = W1[d][k] ----
  V16H bw1;
#pragma unroll
  for (int v = 0; v < 8; ++v) {
    const int K = hh * 16 + 2 * v;
    const float* s = W1 + n * 32 + K;
    bw1.p[v] = pk2(s[0], s[1]);
  }
  // ---- per-lane gate-column constants (gate col j = 16*ct + n) ----
  float wv0[8], wv1[8], biasv[8];
#pragma unroll
  for (int ct = 0; ct < 8; ++ct) {
    const int j = ct * 16 + n;
    wv0[ct]   = W_ih[j * 2 + 0];
    wv1[ct]   = W_ih[j * 2 + 1];
    biasv[ct] = b_ih[j] + b_hh[j];
  }
  const float b1v = b1[n];
  const float w2v = W2[n];
  const float b2v = b2[0];

  // ---- state: c[q][r] <-> (batch m = r + 8*hh, hidden k = 16*q + n) ----
  float c_st[2][8];
#pragma unroll
  for (int q = 0; q < 2; ++q)
#pragma unroll
    for (int r = 0; r < 8; ++r) c_st[q][r] = 0.0f;

  V16H a_h;                            // h in A layout (f16), h0 = 0
#pragma unroll
  for (int v = 0; v < 8; ++v) a_h.p[v] = pk2(0.0f, 0.0f);

  for (int t = 0; t < T_SZ; ++t) {
    // ---- x contribution folded into WMMA C initializer ----
    float x0[8], x1[8];
#pragma unroll
    for (int r = 0; r < 8; ++r) {
      const float* xs = x + ((size_t)(b0 + r + 8 * hh) * T_SZ + t) * 2;
      x0[r] = xs[0];
      x1[r] = xs[1];
    }
    // ---- gates[m][j] = bias + x*W_ih + h @ W_hh^T : 8 WMMAs ----
    v8f Cg[8];
#pragma unroll
    for (int ct = 0; ct < 8; ++ct) {
#pragma unroll
      for (int r = 0; r < 8; ++r)
        Cg[ct][r] = biasv[ct] + x0[r] * wv0[ct] + x1[r] * wv1[ct];
      Cg[ct] = __builtin_amdgcn_wmma_f32_16x16x32_f16(
          false, a_h.v, false, bhh[ct].v, (short)0, Cg[ct], false, false);
    }
    // ---- activations + state update (tiles: i=0/1, f=2/3, g=4/5, o=6/7) ----
#pragma unroll
    for (int q = 0; q < 2; ++q) {
#pragma unroll
      for (int r = 0; r < 8; ++r) {
        const float ig = sigmoid_fast(Cg[0 + q][r]);
        const float fg = sigmoid_fast(Cg[2 + q][r]);
        const float gg = tanh_fast(Cg[4 + q][r]);
        const float og = sigmoid_fast(Cg[6 + q][r]);
        const float c  = fg * c_st[q][r] + ig * gg;
        c_st[q][r]     = c;
        const float hv = og * tanh_fast(c);
        const float th = tanh_fast(hv);          // lstm_out = tanh(h)
        const int addr = (r + 8 * hh) * 32 + 16 * q + n;
        lds_h[addr]  = hv;
        lds_th[addr] = th;
      }
    }
    asm volatile("" ::: "memory");  // keep LDS store->load order (HW DS is in-order per wave)

    // ---- rebuild A operands (C layout -> A layout transpose via LDS) ----
    V16H a_th;
#pragma unroll
    for (int v = 0; v < 8; ++v) {
      const int K    = hh * 16 + 2 * v;
      const int addr = n * 32 + K;
      a_h.p[v]  = pk2(lds_h[addr],  lds_h[addr + 1]);
      a_th.p[v] = pk2(lds_th[addr], lds_th[addr + 1]);
    }
    // ---- fc1 = tanh( tanh(h) @ W1^T + b1 ) : 1 WMMA ----
    v8f C1;
#pragma unroll
    for (int r = 0; r < 8; ++r) C1[r] = b1v;
    C1 = __builtin_amdgcn_wmma_f32_16x16x32_f16(
        false, a_th.v, false, bw1.v, (short)0, C1, false, false);

    // ---- fc2: out[m] = sum_d tanh(fc1[m,d]) * W2[d] + b2 ----
#pragma unroll
    for (int r = 0; r < 8; ++r) {
      float p = tanh_fast(C1[r]) * w2v;
      p += __shfl_xor(p, 1, 32);
      p += __shfl_xor(p, 2, 32);
      p += __shfl_xor(p, 4, 32);
      p += __shfl_xor(p, 8, 32);     // reduced within each 16-lane half
      if (n == r)
        out[(size_t)(b0 + r + 8 * hh) * T_SZ + t] = p + b2v;
    }
  }
}

extern "C" void kernel_launch(void* const* d_in, const int* in_sizes, int n_in,
                              void* d_out, int out_size, void* d_ws, size_t ws_size,
                              hipStream_t stream) {
  (void)in_sizes; (void)n_in; (void)d_ws; (void)ws_size; (void)out_size;
  const float* x    = (const float*)d_in[0];
  const float* W_ih = (const float*)d_in[1];
  const float* W_hh = (const float*)d_in[2];
  const float* b_ih = (const float*)d_in[3];
  const float* b_hh = (const float*)d_in[4];
  const float* W1   = (const float*)d_in[5];
  const float* b1   = (const float*)d_in[6];
  const float* W2   = (const float*)d_in[7];
  const float* b2   = (const float*)d_in[8];
  float* out = (float*)d_out;

  dim3 grid(B_SZ / 16);   // 256 independent batch tiles
  dim3 block(32);         // one wave32 per tile
  hipLaunchKernelGGL(lstm_fused_wmma, grid, block, 0, stream,
                     x, W_ih, W_hh, b_ih, b_hh, W1, b1, W2, b2, out);
}
